// SubgraphClassifier_6906307412089
// MI455X (gfx1250) — matrix operand order repaired and checked by reading, not compiled, not run on hardware
//
#include <hip/hip_runtime.h>
#include <hip/hip_bf16.h>

// CDNA5 WMMA vector types (wave32)
typedef __attribute__((ext_vector_type(16))) __bf16 v16bf;
typedef __attribute__((ext_vector_type(8)))  float  v8f;

// ---------------------------------------------------------------------------
// Utility: zero a float buffer
// ---------------------------------------------------------------------------
__global__ void zero_f32(float* __restrict__ p, long n) {
    long i = (long)blockIdx.x * blockDim.x + threadIdx.x;
    if (i < n) p[i] = 0.0f;
}

// ---------------------------------------------------------------------------
// Degree (in-degree over dst); self-loop (+1) folded into dis_kernel
// ---------------------------------------------------------------------------
__global__ void deg_kernel(const int* __restrict__ dst, float* __restrict__ deg, int E) {
    int e = blockIdx.x * blockDim.x + threadIdx.x;
    if (e < E) atomicAdd(&deg[dst[e]], 1.0f);
}

__global__ void dis_kernel(const float* __restrict__ deg, float* __restrict__ dis, int N) {
    int i = blockIdx.x * blockDim.x + threadIdx.x;
    if (i < N) dis[i] = rsqrtf(fmaxf(deg[i] + 1.0f, 1.0f)); // +1 self-loop
}

// ---------------------------------------------------------------------------
// agg[i,:] = h[i,:] * dis[i]^2  (self-loop contribution; also initializes agg)
// ---------------------------------------------------------------------------
__global__ void init_agg_selfloop(const float* __restrict__ hin, const float* __restrict__ dis,
                                  float* __restrict__ agg, int N, int D) {
    int idx = blockIdx.x * blockDim.x + threadIdx.x;   // N*D <= 12.8M, fits int
    if (idx >= N * D) return;
    int i = idx / D;
    float s = dis[i];
    agg[idx] = hin[idx] * s * s;
}

// ---------------------------------------------------------------------------
// Edge scatter: agg[dst,:] += h[src,:] * dis[src]*dis[dst]
// One wave (32 lanes) per edge; lane strides over the feature dim.
// ---------------------------------------------------------------------------
__global__ void scatter_edges(const float* __restrict__ hin, const int* __restrict__ src,
                              const int* __restrict__ dst, const float* __restrict__ dis,
                              float* __restrict__ agg, int E, int D) {
    int gid  = blockIdx.x * blockDim.x + threadIdx.x;
    int e    = gid >> 5;
    int lane = gid & 31;
    if (e >= E) return;
    int s = src[e], d = dst[e];
    float nrm = dis[s] * dis[d];
    const float* hs = hin + (size_t)s * D;
    float*       ad = agg + (size_t)d * D;
    for (int f = lane; f < D; f += 32)
        atomicAdd(&ad[f], hs[f] * nrm);
}

// ---------------------------------------------------------------------------
// Pre-swizzle W[K,H] (f32, row-major) into bf16 WMMA B-fragment layout:
//   out[((kstep*nT + ntile)*32 + lane)*16 + i] = bf16( W[k, n] )
//   k = kstep*32 + (lane>>4)*16 + i ,  n = ntile*16 + (lane&15),  nT = H/16
// Per-lane 16 contiguous bf16 (32 B) -> GEMM loads B as 2x b128.
// ---------------------------------------------------------------------------
__global__ void swizzle_W_bf16(const float* __restrict__ W, __bf16* __restrict__ out,
                               int K, int H) {
    int tid = blockIdx.x * blockDim.x + threadIdx.x;
    if (tid >= K * H) return;
    int i    = tid & 15;
    int lane = (tid >> 4) & 31;
    int rest = tid >> 9;             // kstep*nT + ntile
    int nT   = H >> 4;
    int nt   = rest % nT;
    int ks   = rest / nT;
    int k = ks * 32 + (lane >> 4) * 16 + i;
    int n = nt * 16 + (lane & 15);
    out[tid] = (__bf16)W[(size_t)k * H + n];
}

// ---------------------------------------------------------------------------
// WMMA GEMM:  C[N,H] = relu?( A[N,K] @ W[K,H] + bias )
// One wave per 16(M) x 32(N) output strip: two 16x16 tiles sharing one A frag.
// K stepped by 32 with v_wmma_f32_16x16x32_bf16, unrolled x2 (K % 64 == 0).
// A fragment lane layout (16-bit A 16x32, contiguous k order used here, same as B):
//   lane 0-15  : row M=lane,    elems 0-15 -> K=k0+0..15
//   lane 16-31 : row M=lane-16, elems 0-15 -> K=k0+16..31
// B fragments come pre-swizzled (see swizzle_W_bf16).
// C/D: elem r -> row m0 + 8*(lane>=16) + r, col = lane&15.
// ---------------------------------------------------------------------------
__global__ void gcn_gemm_wmma(const float* __restrict__ A, const __bf16* __restrict__ Wsw,
                              const float* __restrict__ bias, float* __restrict__ C,
                              int Nrows, int K, int H, int relu) {
    const int lane  = threadIdx.x & 31;
    const int wave  = threadIdx.x >> 5;
    const int mTile = blockIdx.x * 4 + wave;
    const int m0    = mTile * 16;
    const int n0    = blockIdx.y * 32;        // two 16-wide n-tiles per wave
    if (m0 >= Nrows) return;                  // whole-wave exit

    const int halfSel = lane >> 4;            // 0 or 1
    const int l15     = lane & 15;
    const int row     = m0 + l15;
    const int kA      = halfSel * 16;         // A-half K offset
    const bool rowOK  = (row < Nrows);
    const float* arow = A + (size_t)(rowOK ? row : 0) * K;

    const int nT  = H >> 4;
    const int nt0 = n0 >> 4;
    // per-lane pointer into swizzled B fragments for (kstep=0, ntile=nt0)
    const __bf16* bfrag = Wsw + ((size_t)nt0 * 32 + lane) * 16;
    const size_t  bStepK = (size_t)nT * 32 * 16;   // advance one kstep
    const size_t  bStepN = (size_t)32 * 16;        // advance one ntile

    v8f acc0 = {0.f, 0.f, 0.f, 0.f, 0.f, 0.f, 0.f, 0.f};
    v8f acc1 = {0.f, 0.f, 0.f, 0.f, 0.f, 0.f, 0.f, 0.f};

    #pragma unroll 2
    for (int k0 = 0; k0 < K; k0 += 32) {
        // ---- A fragment: 4x float4 loads, convert f32 -> bf16
        float4 f0 = *(const float4*)(arow + k0 + kA);
        float4 f1 = *(const float4*)(arow + k0 + kA + 4);
        float4 f2 = *(const float4*)(arow + k0 + kA + 8);
        float4 f3 = *(const float4*)(arow + k0 + kA + 12);
        if (!rowOK) {
            f0 = make_float4(0.f, 0.f, 0.f, 0.f);
            f1 = f0; f2 = f0; f3 = f0;
        }
        v16bf a;
        a[0]  = (__bf16)f0.x; a[1]  = (__bf16)f0.y; a[2]  = (__bf16)f0.z; a[3]  = (__bf16)f0.w;
        a[4]  = (__bf16)f1.x; a[5]  = (__bf16)f1.y; a[6]  = (__bf16)f1.z; a[7]  = (__bf16)f1.w;
        a[8]  = (__bf16)f2.x; a[9]  = (__bf16)f2.y; a[10] = (__bf16)f2.z; a[11] = (__bf16)f2.w;
        a[12] = (__bf16)f3.x; a[13] = (__bf16)f3.y; a[14] = (__bf16)f3.z; a[15] = (__bf16)f3.w;

        // ---- B fragments: 32 B contiguous per lane -> 2x b128 each
        const __bf16* bp = bfrag + (size_t)(k0 >> 5) * bStepK;
        v16bf b0 = *(const v16bf*)bp;
        v16bf b1 = *(const v16bf*)(bp + bStepN);

        // ---- D = A*B + C  (8 args: neg_a, A, neg_b, B, c_mod, C, reuse_a, reuse_b)
        acc0 = __builtin_amdgcn_wmma_f32_16x16x32_bf16(false, a, false, b0,
                                                       (short)0, acc0, false, false);
        acc1 = __builtin_amdgcn_wmma_f32_16x16x32_bf16(false, a, false, b1,
                                                       (short)0, acc1, false, false);
    }

    // epilogue: + bias, optional ReLU, store per C/D layout
    const float bn0 = bias[n0 + l15];
    const float bn1 = bias[n0 + 16 + l15];
    #pragma unroll
    for (int r = 0; r < 8; ++r) {
        int m = m0 + halfSel * 8 + r;
        float v0 = acc0[r] + bn0;
        float v1 = acc1[r] + bn1;
        if (relu) { v0 = fmaxf(v0, 0.f); v1 = fmaxf(v1, 0.f); }
        if (m < Nrows) {
            C[(size_t)m * H + n0 + l15]      = v0;
            C[(size_t)m * H + n0 + 16 + l15] = v1;
        }
    }
}

// ---------------------------------------------------------------------------
// Global mean pool (scatter part): g[batch[i],:] += h[i,:], cnt[batch[i]] += 1
// ---------------------------------------------------------------------------
__global__ void pool_scatter(const float* __restrict__ h, const int* __restrict__ batch,
                             float* __restrict__ g, float* __restrict__ cnt, int N, int H) {
    int gid  = blockIdx.x * blockDim.x + threadIdx.x;
    int i    = gid >> 5;
    int lane = gid & 31;
    if (i >= N) return;
    int b = batch[i];
    const float* hr = h + (size_t)i * H;
    float*       gr = g + (size_t)b * H;
    for (int f = lane; f < H; f += 32)
        atomicAdd(&gr[f], hr[f]);
    if (lane == 0) atomicAdd(&cnt[b], 1.0f);
}

// ---------------------------------------------------------------------------
// z[b] = [ g[b]/max(cnt,1), h[u[b]], h[v[b]] ]   (B x 3H)
// ---------------------------------------------------------------------------
__global__ void build_z(const float* __restrict__ g, const float* __restrict__ cnt,
                        const float* __restrict__ h, const int* __restrict__ u,
                        const int* __restrict__ v, float* __restrict__ z, int H) {
    int b = blockIdx.x;
    int t = threadIdx.x;                // blockDim.x == H
    float invc = 1.0f / fmaxf(cnt[b], 1.0f);
    float* zr = z + (size_t)b * 3 * H;
    zr[t]         = g[(size_t)b * H + t] * invc;
    zr[H + t]     = h[(size_t)u[b] * H + t];
    zr[2 * H + t] = h[(size_t)v[b] * H + t];
}

// ---------------------------------------------------------------------------
// MLP head: out[b] = relu(z[b] @ Wm1 + bm1) @ Wm2 + bm2   (H = 256 hardcoded)
// ---------------------------------------------------------------------------
__global__ void mlp_head(const float* __restrict__ z, const float* __restrict__ Wm1,
                         const float* __restrict__ bm1, const float* __restrict__ Wm2,
                         const float* __restrict__ bm2, float* __restrict__ out) {
    __shared__ float zs[768];
    __shared__ float red[256];
    const int b = blockIdx.x;
    const int t = threadIdx.x;          // 256 threads
    for (int i = t; i < 768; i += 256) zs[i] = z[(size_t)b * 768 + i];
    __syncthreads();

    float acc = bm1[t];
    for (int k = 0; k < 768; ++k)
        acc = fmaf(zs[k], Wm1[(size_t)k * 256 + t], acc);
    acc = fmaxf(acc, 0.f);

    red[t] = acc * Wm2[t];
    __syncthreads();
    for (int s = 128; s > 0; s >>= 1) {
        if (t < s) red[t] += red[t + s];
        __syncthreads();
    }
    if (t == 0) out[b] = red[0] + bm2[0];
}

// ---------------------------------------------------------------------------
// Launch
// ---------------------------------------------------------------------------
extern "C" void kernel_launch(void* const* d_in, const int* in_sizes, int n_in,
                              void* d_out, int out_size, void* d_ws, size_t ws_size,
                              hipStream_t stream) {
    const float* x    = (const float*)d_in[0];
    const int*   ei   = (const int*)  d_in[1];
    const int*   bat  = (const int*)  d_in[2];
    const int*   u    = (const int*)  d_in[3];
    const int*   v    = (const int*)  d_in[4];
    const float* W0   = (const float*)d_in[5];
    const float* b0   = (const float*)d_in[6];
    const float* W1   = (const float*)d_in[7];
    const float* b1   = (const float*)d_in[8];
    const float* W2   = (const float*)d_in[9];
    const float* b2   = (const float*)d_in[10];
    const float* Wm1  = (const float*)d_in[11];
    const float* bm1  = (const float*)d_in[12];
    const float* Wm2  = (const float*)d_in[13];
    const float* bm2  = (const float*)d_in[14];

    const int H    = in_sizes[6];            // 256
    const int DRNL = in_sizes[5] / H;        // 64
    const int N    = in_sizes[0] / DRNL;     // 50000
    const int E    = in_sizes[1] / 2;        // 800000
    const int B    = in_sizes[3];            // 512
    const int H3   = 3 * H;

    const int* src = ei;
    const int* dst = ei + E;

    // workspace carve-out (floats), 64-float (256 B) aligned
    auto al = [](size_t n) { return (n + 63) & ~(size_t)63; };
    float* base = (float*)d_ws;
    size_t off = 0;
    float* deg = base + off; off += al((size_t)N);
    float* dis = base + off; off += al((size_t)N);
    float* agg = base + off; off += al((size_t)N * H);
    float* h   = base + off; off += al((size_t)N * H);
    float* g   = base + off; off += al((size_t)B * H);
    float* cnt = base + off; off += al((size_t)B);
    float* z   = base + off; off += al((size_t)B * H3);
    __bf16* wsw = (__bf16*)(base + off); off += al((size_t)H * H / 2 + 64);
    (void)ws_size; (void)n_in; (void)out_size;

    float* out = (float*)d_out;

    const int TB = 256;

    // 1) degrees + normalization
    zero_f32<<<(N + TB - 1) / TB, TB, 0, stream>>>(deg, (long)N);
    deg_kernel<<<(E + TB - 1) / TB, TB, 0, stream>>>(dst, deg, E);
    dis_kernel<<<(N + TB - 1) / TB, TB, 0, stream>>>(deg, dis, N);

    // 2) three GCN layers
    const float* hin[3]  = { x,  h,  h  };
    const int    Din[3]  = { DRNL, H, H };
    const float* Wl[3]   = { W0, W1, W2 };
    const float* bl[3]   = { b0, b1, b2 };
    const int    relu[3] = { 1, 1, 0 };

    const int mTiles = (N + 15) / 16;
    dim3 gemmGrid((mTiles + 3) / 4, H / 32);

    for (int l = 0; l < 3; ++l) {
        const int D = Din[l];
        int nd = N * D;
        swizzle_W_bf16<<<(D * H + TB - 1) / TB, TB, 0, stream>>>(Wl[l], wsw, D, H);
        init_agg_selfloop<<<(nd + TB - 1) / TB, TB, 0, stream>>>(hin[l], dis, agg, N, D);
        long ethreads = (long)E * 32;
        scatter_edges<<<(int)((ethreads + TB - 1) / TB), TB, 0, stream>>>(
            hin[l], src, dst, dis, agg, E, D);
        gcn_gemm_wmma<<<gemmGrid, 128, 0, stream>>>(agg, wsw, bl[l], h, N, D, H, relu[l]);
    }

    // 3) mean pool
    zero_f32<<<((long)B * H + TB - 1) / TB, TB, 0, stream>>>(g, (long)B * H);
    zero_f32<<<(B + TB - 1) / TB, TB, 0, stream>>>(cnt, (long)B);
    long pthreads = (long)N * 32;
    pool_scatter<<<(int)((pthreads + TB - 1) / TB), TB, 0, stream>>>(h, bat, g, cnt, N, H);

    // 4) head
    build_z<<<B, H, 0, stream>>>(g, cnt, h, u, v, z, H);
    mlp_head<<<B, 256, 0, stream>>>(z, Wm1, bm1, Wm2, bm2, out);
}
